// TripletCriterion_68899865362859
// MI455X (gfx1250) — matrix operand order repaired and checked by reading, not compiled, not run on hardware
//
#include <hip/hip_runtime.h>
#include <hip/hip_bf16.h>
#include <math.h>

// CDNA5 / gfx1250: wave32, WMMA 16x16x32 bf16 (f32 accumulate).
typedef __attribute__((ext_vector_type(16))) __bf16 v16bf;
typedef __attribute__((ext_vector_type(8)))  float  v8f;

#define D       128   // feature dim (fixed by the reference)
#define WAVES   4     // waves per workgroup
#define ROWS_WG (WAVES * 16)
#define TILE_N  32    // staged P columns per iteration (two 16-wide B tiles)

// ---------------------------------------------------------------- p2 = ||pos_t||^2
__global__ void p2_kernel(const float* __restrict__ pos, float* __restrict__ p2, int n) {
    int s = blockIdx.x * blockDim.x + threadIdx.x;
    if (s >= n) return;
    const float4* p = (const float4*)(pos + (size_t)s * D);
    float acc = 0.f;
#pragma unroll
    for (int i = 0; i < D / 4; ++i) {
        float4 v = p[i];
        acc += v.x * v.x + v.y * v.y + v.z * v.z + v.w * v.w;
    }
    p2[s] = acc;
}

// ------------------------------------------- fused block-diag GEMM + diag-mask + argmin
// score(s,t) = p2[t] - 2 * (a_s . p_t)   (a2[s] is constant per row -> dropped)
// cross computed with bf16 hi/lo split: a.p ~= ah.bh + ah.bl + al.bh  (f32 accum)
__global__ __launch_bounds__(WAVES * 32)
void hardneg_kernel(const float* __restrict__ anc, const float* __restrict__ pos,
                    const float* __restrict__ p2, int* __restrict__ hn, int S) {
    const int crop = blockIdx.y;
    const int wave = threadIdx.x >> 5;
    const int lane = threadIdx.x & 31;
    const int m    = lane & 15;    // row (A) / col (B) within 16-tile
    const int h    = lane >> 4;    // half-wave selector
    const int rowbase = blockIdx.x * ROWS_WG + wave * 16;  // row base within crop

    __shared__ __bf16 p_hi[TILE_N * D];   // 8 KB
    __shared__ __bf16 p_lo[TILE_N * D];   // 8 KB

    // ---- A stripe: load 16 rows, split into bf16 hi/lo fragments, keep in VGPRs.
    // A 16x32 in contiguous k order (same as B): lanes 0-15 -> M=lane; elem e -> K = kb + h*16 + e.
    v16bf ahi[4], alo[4];
    {
        const float* arow = anc + (size_t)(crop * S + rowbase + m) * D;
#pragma unroll
        for (int kt = 0; kt < 4; ++kt) {
            const int kb = kt * 32;
#pragma unroll
            for (int e = 0; e < 16; ++e) {
                int K = kb + h * 16 + e;
                float x = arow[K];
                __bf16 hi = (__bf16)x;
                ahi[kt][e] = hi;
                alo[kt][e] = (__bf16)(x - (float)hi);
            }
        }
    }

    float minv[8];
    int   mini[8];
#pragma unroll
    for (int r = 0; r < 8; ++r) { minv[r] = 3.4e38f; mini[r] = 0; }

    const float* pcrop = pos + (size_t)crop * S * D;
    for (int t0 = 0; t0 < S; t0 += TILE_N) {
        __syncthreads();
        // Stage P tile (32 consecutive rows are contiguous) as bf16 hi/lo in LDS.
        const float* pbase = pcrop + (size_t)t0 * D;
        for (int i = threadIdx.x; i < TILE_N * D; i += WAVES * 32) {
            float x = pbase[i];
            __bf16 hi = (__bf16)x;
            p_hi[i] = hi;
            p_lo[i] = (__bf16)(x - (float)hi);
        }
        __syncthreads();

        // Prefetch next tile into cache while the XDL pipe chews on this one.
        // 128 threads x 128B stride covers the next 16 KB tile exactly.
        if (t0 + TILE_N < S)
            __builtin_prefetch(pcrop + (size_t)(t0 + TILE_N) * D + threadIdx.x * 32, 0, 3);

        // B 32x16 in contiguous k order (same as A): lane n holds column n; elem e -> K = kb + h*16 + e.
        // Two 16-wide B tiles -> two independent WMMA accumulation chains (ILP).
        v8f acc0 = {}, acc1 = {};
#pragma unroll
        for (int kt = 0; kt < 4; ++kt) {
            v16bf bh0, bl0, bh1, bl1;
            const int base0 = m * D + kt * 32 + h * 16;
            const int base1 = (m + 16) * D + kt * 32 + h * 16;
#pragma unroll
            for (int e = 0; e < 16; ++e) {
                bh0[e] = p_hi[base0 + e]; bl0[e] = p_lo[base0 + e];
                bh1[e] = p_hi[base1 + e]; bl1[e] = p_lo[base1 + e];
            }
            acc0 = __builtin_amdgcn_wmma_f32_16x16x32_bf16(false, ahi[kt], false, bh0, (short)0, acc0, false, false);
            acc1 = __builtin_amdgcn_wmma_f32_16x16x32_bf16(false, ahi[kt], false, bh1, (short)0, acc1, false, false);
            acc0 = __builtin_amdgcn_wmma_f32_16x16x32_bf16(false, ahi[kt], false, bl0, (short)0, acc0, false, false);
            acc1 = __builtin_amdgcn_wmma_f32_16x16x32_bf16(false, ahi[kt], false, bl1, (short)0, acc1, false, false);
            acc0 = __builtin_amdgcn_wmma_f32_16x16x32_bf16(false, alo[kt], false, bh0, (short)0, acc0, false, false);
            acc1 = __builtin_amdgcn_wmma_f32_16x16x32_bf16(false, alo[kt], false, bh1, (short)0, acc1, false, false);
        }

        // Epilogue: C/D layout -> elem r of lane l is (row = rowbase + r + 8h, col).
        const int   col0 = t0 + m;
        const int   col1 = t0 + 16 + m;
        const float pv0  = p2[crop * S + col0];
        const float pv1  = p2[crop * S + col1];
#pragma unroll
        for (int r = 0; r < 8; ++r) {
            const int row = rowbase + r + 8 * h;
            float s0 = pv0 - 2.0f * acc0[r];
            float s1 = pv1 - 2.0f * acc1[r];
            if (row == col0) s0 = 3.4e38f;           // fill_diagonal_(POS_DIST)
            if (row == col1) s1 = 3.4e38f;
            if (s0 < minv[r]) { minv[r] = s0; mini[r] = col0; }   // strict < keeps first
            if (s1 < minv[r]) { minv[r] = s1; mini[r] = col1; }   // col1 > col0 -> order ok
        }
    }

    // Min+index reduce across the 16 lanes that share each output row (tie -> lower idx).
#pragma unroll
    for (int r = 0; r < 8; ++r) {
#pragma unroll
        for (int off = 8; off; off >>= 1) {
            float ov = __shfl_xor(minv[r], off, 16);
            int   oi = __shfl_xor(mini[r], off, 16);
            if (ov < minv[r] || (ov == minv[r] && oi < mini[r])) { minv[r] = ov; mini[r] = oi; }
        }
    }
    if (m == 0) {
#pragma unroll
        for (int r = 0; r < 8; ++r)
            hn[crop * S + rowbase + r + 8 * h] = mini[r];
    }
}

// ---------------------------------------------------------------- triplet loss partials
__global__ void loss_kernel(const float* __restrict__ anc, const float* __restrict__ pos,
                            const int* __restrict__ hn, float* __restrict__ partials,
                            int n, int S) {
    int s = blockIdx.x * blockDim.x + threadIdx.x;
    float v = 0.f;
    if (s < n) {
        int c = s / S;
        const float4* a = (const float4*)(anc + (size_t)s * D);
        const float4* p = (const float4*)(pos + (size_t)s * D);
        const float4* q = (const float4*)(pos + (size_t)(c * S + hn[s]) * D);
        float sap = 0.f, san = 0.f;
#pragma unroll 8
        for (int i = 0; i < D / 4; ++i) {
            float4 av = a[i], pv = p[i], qv = q[i];
            float d0 = av.x - pv.x + 1e-6f, d1 = av.y - pv.y + 1e-6f;
            float d2 = av.z - pv.z + 1e-6f, d3 = av.w - pv.w + 1e-6f;
            sap += d0 * d0 + d1 * d1 + d2 * d2 + d3 * d3;
            float e0 = av.x - qv.x + 1e-6f, e1 = av.y - qv.y + 1e-6f;
            float e2 = av.z - qv.z + 1e-6f, e3 = av.w - qv.w + 1e-6f;
            san += e0 * e0 + e1 * e1 + e2 * e2 + e3 * e3;
        }
        v = fmaxf(sqrtf(sap) - sqrtf(san) + 1.0f, 0.f);
    }
    __shared__ float red[256];
    red[threadIdx.x] = v;
    __syncthreads();
    for (int off = blockDim.x >> 1; off; off >>= 1) {
        if ((int)threadIdx.x < off) red[threadIdx.x] += red[threadIdx.x + off];
        __syncthreads();
    }
    if (threadIdx.x == 0) partials[blockIdx.x] = red[0];
}

// Deterministic serial final sum (no float atomics -> bit-identical replays).
__global__ void finalize_kernel(const float* __restrict__ partials, float* __restrict__ out,
                                int nb, float inv_n) {
    if (threadIdx.x == 0) {
        float s = 0.f;
        for (int i = 0; i < nb; ++i) s += partials[i];
        out[0] = s * inv_n;
    }
}

// ------------------------------------------------------------------------- launcher
extern "C" void kernel_launch(void* const* d_in, const int* in_sizes, int n_in,
                              void* d_out, int out_size, void* d_ws, size_t ws_size,
                              hipStream_t stream) {
    const float* anc = (const float*)d_in[0];
    const float* pos = (const float*)d_in[1];
    const int n = in_sizes[0] / D;   // 16384
    const int C = in_sizes[2];       // 8 crops
    const int S = n / C;             // 2048 per crop

    float* p2       = (float*)d_ws;
    int*   hn       = (int*)((char*)d_ws + (size_t)n * sizeof(float));
    float* partials = (float*)((char*)d_ws + (size_t)n * (sizeof(float) + sizeof(int)));

    p2_kernel<<<(n + 255) / 256, 256, 0, stream>>>(pos, p2, n);

    dim3 grid(S / ROWS_WG, C);       // (32, 8) = 256 workgroups of 4 waves
    hardneg_kernel<<<grid, WAVES * 32, 0, stream>>>(anc, pos, p2, hn, S);

    const int nb = (n + 255) / 256;
    loss_kernel<<<nb, 256, 0, stream>>>(anc, pos, hn, partials, n, S);
    finalize_kernel<<<1, 32, 0, stream>>>(partials, (float*)d_out, nb, 1.0f / (float)n);
}
